// EntityMultiAttnQMixer_44092134261285
// MI455X (gfx1250) — matrix operand order repaired and checked by reading, not compiled, not run on hardware
//
#include <hip/hip_runtime.h>
#include <hip/hip_bf16.h>
#include <math.h>

// ---------------------------------------------------------------------------
// EntityMultiAttnQMixer on MI455X (gfx1250, wave32, WMMA).
// One 256-thread workgroup (8 waves) per sample (bs = 64*256 = 16384).
// All GEMMs use v_wmma_f32_16x16x32_f16; weights pre-converted to f16 and
// pre-swizzled into B-fragment lane order in d_ws by a small pre-kernel.
// ---------------------------------------------------------------------------

typedef __attribute__((ext_vector_type(16))) _Float16 v16h;
typedef __attribute__((ext_vector_type(8)))  float    v8f;

#define WMMA(a, b, c) __builtin_amdgcn_wmma_f32_16x16x32_f16( \
    false, (a), false, (b), (short)0, (c), false, false)

struct WPtrs  { const float* w[24]; };            // [branch*6 + mat] weights (f32, 64x64)
struct KParams {
  const float* qs;                                // (bs,16)
  const float* ents;                              // (bs,32,64)
  const unsigned char* mask;                      // (bs,32) bool
  const float* bias[24];                          // [branch*6 + mat] biases (f32, 64)
  const _Float16* wf;                             // d_ws fragments [24][8][512]
  float* out;                                     // (bs,)
};

// A-fragment (16x32 f16, M x K) from row-major f16 LDS, contiguous k order (as load_B):
// lanes 0-15: M=lane, K = 0..15; lanes 16-31: M=lane-16, K = 16..31
__device__ __forceinline__ v16h load_A(const _Float16* s, int ld, int row0, int k0) {
  int lane = threadIdx.x & 31;
  int m  = lane & 15;
  int kb = (lane < 16) ? 0 : 16;
  v16h a;
#pragma unroll
  for (int j = 0; j < 16; ++j) {
    int k = kb + j;
    a[j] = s[(row0 + m) * ld + k0 + k];
  }
  return a;
}

// Same but only K<16 is real data (attention logits, hd=16): pad K 16..31 = 0.
__device__ __forceinline__ v16h load_A_pad16(const _Float16* s, int ld, int row0, int kbase) {
  int lane = threadIdx.x & 31;
  int m  = lane & 15;
  int kb = (lane < 16) ? 0 : 16;
  v16h a;
#pragma unroll
  for (int j = 0; j < 16; ++j) {
    int k = kb + j;
    a[j] = (k < 16) ? s[(row0 + m) * ld + kbase + k] : (_Float16)0.0f;
  }
  return a;
}

// B-fragment (32x16 f16, K x N) from row-major f16 LDS:
// lanes 0-15: N=lane, K=0..15 in halves 0..15; lanes 16-31: N=lane-16, K=16..31
__device__ __forceinline__ v16h load_B(const _Float16* s, int ld, int k0, int n0) {
  int lane = threadIdx.x & 31;
  int n  = lane & 15;
  int kb = (lane < 16) ? 0 : 16;
  v16h b;
#pragma unroll
  for (int j = 0; j < 16; ++j) b[j] = s[(k0 + kb + j) * ld + n0 + n];
  return b;
}

// B = K^T for logits: B(k,n) = Kmat[n0+n][kbase+k], valid k<16, pad rest 0.
__device__ __forceinline__ v16h load_Bt_pad16(const _Float16* s, int ld, int n0, int kbase) {
  int lane = threadIdx.x & 31;
  int n  = lane & 15;
  int kb = (lane < 16) ? 0 : 16;
  v16h b;
#pragma unroll
  for (int j = 0; j < 16; ++j) {
    int k = kb + j;
    b[j] = (k < 16) ? s[(n0 + n) * ld + kbase + k] : (_Float16)0.0f;
  }
  return b;
}

// ---------------------------------------------------------------------------
// Pre-kernel: f32 weights -> f16 B-fragments in d_ws.
// Layout: frag element index e = ((mi*8 + ks*4 + nt)*32 + lane)*16 + j
// value  = W[ (ks*32 + ((lane<16)?0:16) + j) * 64 + nt*16 + (lane&15) ]
// ---------------------------------------------------------------------------
__global__ __launch_bounds__(256) void qmix_convert_weights(WPtrs p, _Float16* out) {
  int e = blockIdx.x * 256 + threadIdx.x;
  if (e >= 24 * 8 * 512) return;
  int j    = e & 15;
  int lane = (e >> 4) & 31;
  int f    = (e >> 9) & 7;
  int mi   = e >> 12;
  int ks   = f >> 2, nt = f & 3;
  int kb   = (lane < 16) ? 0 : 16;
  int k    = ks * 32 + kb + j;
  int col  = nt * 16 + (lane & 15);
  out[e] = (_Float16)p.w[mi][k * 64 + col];
}

// ---------------------------------------------------------------------------
// Main kernel: one workgroup per sample.
// ---------------------------------------------------------------------------
__global__ __launch_bounds__(256) void qmix_kernel(KParams p) {
  __shared__ _Float16 Ef[32 * 64];      // entities f16
  __shared__ _Float16 Xs[32 * 64];      // relu(emb)
  __shared__ _Float16 Qs[16 * 64];      // q (agents only)
  __shared__ _Float16 Ks_[32 * 64];     // k
  __shared__ _Float16 Vs[32 * 64];      // v
  __shared__ _Float16 Ps[4 * 16 * 32];  // attn probs per head
  __shared__ _Float16 AOs[16 * 64];     // attn output (agents)
  __shared__ _Float16 Ys[16 * 64];      // masked o-proj output
  __shared__ float    Zs[4][16 * 64];   // hyper outputs per branch (f32)
  __shared__ float    qs_s[16];
  __shared__ unsigned char em_s[32];
  __shared__ float    m1s[16], s1s[16], m2s[16], s2s[16], r3s[16];
  __shared__ float    part[64];

  const int tid  = threadIdx.x;
  const int lane = tid & 31;
  const int wave = tid >> 5;
  const int samp = blockIdx.x;
  const int nlo  = lane & 15;
  const int rb   = (lane < 16) ? 0 : 8;

  const float* ent = p.ents + (size_t)samp * 2048;
  for (int i = tid; i < 2048; i += 256) Ef[i] = (_Float16)ent[i];
  if (tid < 16) qs_s[tid] = p.qs[(size_t)samp * 16 + tid];
  if (tid < 32) em_s[tid] = p.mask[(size_t)samp * 32 + tid];
  __syncthreads();

  for (int br = 0; br < 4; ++br) {
    const _Float16* wfb = p.wf + (size_t)br * 6 * 8 * 512;

    // ---- stage 1: X = relu(E @ Wemb + b), 32x64, 8 tiles, one per wave ----
    {
      int mt = wave >> 2, nt = wave & 3;
      v8f acc = {};
#pragma unroll
      for (int ks2 = 0; ks2 < 2; ++ks2) {
        v16h a = load_A(Ef, 64, mt * 16, ks2 * 32);
        v16h b = *(const v16h*)(wfb + (0 * 8 + ks2 * 4 + nt) * 512 + lane * 16);
        acc = WMMA(a, b, acc);
      }
      float bb = p.bias[br * 6 + 0][nt * 16 + nlo];
#pragma unroll
      for (int i = 0; i < 8; ++i) {
        float v = acc[i] + bb;
        Xs[(mt * 16 + rb + i) * 64 + nt * 16 + nlo] = (_Float16)(v > 0.0f ? v : 0.0f);
      }
    }
    __syncthreads();

    // ---- stage 2: Q (16x64), K (32x64), V (32x64) = X @ W + b: 20 tiles ----
    for (int t = wave; t < 20; t += 8) {
      int mat, mt, nt; _Float16* dst;
      if (t < 4)       { mat = 1; mt = 0;             nt = t;            dst = Qs;  }
      else if (t < 12) { int u = t - 4;  mat = 2; mt = u >> 2; nt = u & 3; dst = Ks_; }
      else             { int u = t - 12; mat = 3; mt = u >> 2; nt = u & 3; dst = Vs;  }
      v8f acc = {};
#pragma unroll
      for (int ks2 = 0; ks2 < 2; ++ks2) {
        v16h a = load_A(Xs, 64, mt * 16, ks2 * 32);
        v16h b = *(const v16h*)(wfb + (mat * 8 + ks2 * 4 + nt) * 512 + lane * 16);
        acc = WMMA(a, b, acc);
      }
      float bb = p.bias[br * 6 + mat][nt * 16 + nlo];
#pragma unroll
      for (int i = 0; i < 8; ++i)
        dst[(mt * 16 + rb + i) * 64 + nt * 16 + nlo] = (_Float16)(acc[i] + bb);
    }
    __syncthreads();

    // ---- stages 3+4: per-head logits, masked softmax, P @ V (waves 0-3) ----
    if (wave < 4) {
      int h = wave;
      v16h aq  = load_A_pad16(Qs, 64, 0, h * 16);
      v16h bk0 = load_Bt_pad16(Ks_, 64, 0,  h * 16);
      v16h bk1 = load_Bt_pad16(Ks_, 64, 16, h * 16);
      v8f l0 = {}; l0 = WMMA(aq, bk0, l0);
      v8f l1 = {}; l1 = WMMA(aq, bk1, l1);
      bool k0m = em_s[nlo] != 0;
      bool k1m = em_s[16 + nlo] != 0;
#pragma unroll
      for (int i = 0; i < 8; ++i) {
        float a0 = l0[i] * 0.25f; if (k0m) a0 = -1.0e9f;   // /sqrt(16), then mask
        float a1 = l1[i] * 0.25f; if (k1m) a1 = -1.0e9f;
        float mx = fmaxf(a0, a1);
        mx = fmaxf(mx, __shfl_xor(mx, 1, 32));
        mx = fmaxf(mx, __shfl_xor(mx, 2, 32));
        mx = fmaxf(mx, __shfl_xor(mx, 4, 32));
        mx = fmaxf(mx, __shfl_xor(mx, 8, 32));
        float e0 = __expf(a0 - mx), e1 = __expf(a1 - mx);
        float sm = e0 + e1;
        sm += __shfl_xor(sm, 1, 32);
        sm += __shfl_xor(sm, 2, 32);
        sm += __shfl_xor(sm, 4, 32);
        sm += __shfl_xor(sm, 8, 32);
        float inv = 1.0f / sm;
        int row = rb + i;
        Ps[h * 512 + row * 32 + nlo]      = (_Float16)(e0 * inv);
        Ps[h * 512 + row * 32 + 16 + nlo] = (_Float16)(e1 * inv);
      }
      // out_h(16x16) = P_h(16x32) @ V[:, 16h:16h+16]  (same wave; DS in-order)
      v16h ap = load_A(Ps + h * 512, 32, 0, 0);
      v16h bv = load_B(Vs, 64, 0, h * 16);
      v8f o = {}; o = WMMA(ap, bv, o);
#pragma unroll
      for (int i = 0; i < 8; ++i)
        AOs[(rb + i) * 64 + h * 16 + nlo] = (_Float16)o[i];
    }
    __syncthreads();

    // ---- stage 5: Y = mask(AO @ Wo + b) (16x64), waves 0-3 ----
    if (wave < 4) {
      int nt = wave;
      v8f acc = {};
#pragma unroll
      for (int ks2 = 0; ks2 < 2; ++ks2) {
        v16h a = load_A(AOs, 64, 0, ks2 * 32);
        v16h b = *(const v16h*)(wfb + (4 * 8 + ks2 * 4 + nt) * 512 + lane * 16);
        acc = WMMA(a, b, acc);
      }
      float bb = p.bias[br * 6 + 4][nt * 16 + nlo];
#pragma unroll
      for (int i = 0; i < 8; ++i) {
        int r = rb + i;
        float v = acc[i] + bb;
        if (em_s[r]) v = 0.0f;                    // agent mask zeroes y rows
        Ys[r * 64 + nt * 16 + nlo] = (_Float16)v;
      }
    }
    __syncthreads();

    // ---- stage 6: Z[br] = Y @ Whyper + b (16x64, f32), waves 0-3 ----
    if (wave < 4) {
      int nt = wave;
      v8f acc = {};
#pragma unroll
      for (int ks2 = 0; ks2 < 2; ++ks2) {
        v16h a = load_A(Ys, 64, 0, ks2 * 32);
        v16h b = *(const v16h*)(wfb + (5 * 8 + ks2 * 4 + nt) * 512 + lane * 16);
        acc = WMMA(a, b, acc);
      }
      float bb = p.bias[br * 6 + 5][nt * 16 + nlo];
#pragma unroll
      for (int i = 0; i < 8; ++i)
        Zs[br][(rb + i) * 64 + nt * 16 + nlo] = acc[i] + bb;
    }
    __syncthreads();
  }

  // ---- combine: softmaxes / means / elu / final dot ----
  if (tid < 16) {
    int a = tid;
    float mx1 = -3.4e38f, mx2 = -3.4e38f;
    for (int c = 0; c < 64; ++c) {
      mx1 = fmaxf(mx1, Zs[0][a * 64 + c]);
      mx2 = fmaxf(mx2, Zs[2][a * 64 + c]);
    }
    float s1 = 0.f, s2 = 0.f, r3 = 0.f;
    for (int c = 0; c < 64; ++c) {
      s1 += __expf(Zs[0][a * 64 + c] - mx1);
      s2 += __expf(Zs[2][a * 64 + c] - mx2);
      r3 += Zs[3][a * 64 + c];
    }
    m1s[a] = mx1; s1s[a] = 1.0f / s1;
    m2s[a] = mx2; s2s[a] = 1.0f / s2;
    r3s[a] = em_s[a] ? 0.0f : r3;            // b_2 numerator, masked rows zero
  }
  __syncthreads();
  if (tid < 64) {
    int d = tid;
    float b1 = 0.f, hn = 0.f, w2 = 0.f;
    for (int a = 0; a < 16; ++a) {
      // w_1 softmax includes masked rows (their z = hyper bias)
      hn += qs_s[a] * __expf(Zs[0][a * 64 + d] - m1s[a]) * s1s[a];
      if (!em_s[a]) {
        b1 += Zs[1][a * 64 + d];
        w2 += __expf(Zs[2][a * 64 + d] - m2s[a]) * s2s[a];
      }
    }
    float hv = hn + b1 * (1.0f / 16.0f);     // qs @ w_1 + b_1
    hv = hv > 0.0f ? hv : (__expf(hv) - 1.0f);  // elu
    part[d] = hv * (w2 * (1.0f / 16.0f));    // h * w_2
  }
  __syncthreads();
  if (tid == 0) {
    float b2 = 0.f;
    for (int a = 0; a < 16; ++a) b2 += r3s[a];
    float q = b2 * (1.0f / 1024.0f);         // mean over 16*64
    for (int d = 0; d < 64; ++d) q += part[d];
    p.out[samp] = q;
  }
}

// ---------------------------------------------------------------------------
// d_in order (jax sorted tree-flatten): 0 agent_qs, 1 entities, 2 entity_mask,
// then params sorted: attn_b1{k,o,q,v}(w,b)=3..10, attn_b2=11..18,
// attn_w1=19..26, attn_w2=27..34, emb_b1=35/36, emb_b2=37/38, emb_w1=39/40,
// emb_w2=41/42, hyper_b1=43/44, hyper_b2=45/46, hyper_w1=47/48, hyper_w2=49/50
// ---------------------------------------------------------------------------
extern "C" void kernel_launch(void* const* d_in, const int* in_sizes, int n_in,
                              void* d_out, int out_size, void* d_ws, size_t ws_size,
                              hipStream_t stream) {
  (void)in_sizes; (void)n_in; (void)out_size; (void)ws_size;
  // weight index per [branch][mat]; mat: 0=emb 1=q 2=k 3=v 4=o 5=hyper; bias = idx+1
  static const int wi[4][6] = {
      {39, 23, 19, 25, 21, 47},   // branch 0: w1
      {35,  7,  3,  9,  5, 43},   // branch 1: b1
      {41, 31, 27, 33, 29, 49},   // branch 2: w2
      {37, 15, 11, 17, 13, 45},   // branch 3: b2
  };
  WPtrs wp;
  KParams kp;
  for (int b = 0; b < 4; ++b)
    for (int m = 0; m < 6; ++m) {
      wp.w[b * 6 + m]    = (const float*)d_in[wi[b][m]];
      kp.bias[b * 6 + m] = (const float*)d_in[wi[b][m] + 1];
    }
  kp.qs   = (const float*)d_in[0];
  kp.ents = (const float*)d_in[1];
  kp.mask = (const unsigned char*)d_in[2];
  kp.wf   = (const _Float16*)d_ws;
  kp.out  = (float*)d_out;

  _Float16* wf = (_Float16*)d_ws;                 // needs 24*8*512*2 = 192 KiB
  qmix_convert_weights<<<384, 256, 0, stream>>>(wp, wf);   // 98304 elements
  qmix_kernel<<<64 * 256, 256, 0, stream>>>(kp);           // one WG per sample
}
